// ConvCaps_77154792506129
// MI455X (gfx1250) — matrix operand order repaired and checked by reading, not compiled, not run on hardware
//
#include <hip/hip_runtime.h>

typedef _Float16 half8 __attribute__((ext_vector_type(8)));
typedef _Float16 v16h  __attribute__((ext_vector_type(16)));
typedef float    v8f   __attribute__((ext_vector_type(8)));
typedef float    f4    __attribute__((ext_vector_type(4)));

#define NB 32
#define NH 64
#define NW 64
#define NCHAN 64          // C*D = NCH*DIM = 64
#define PC 66             // 64 cols + halo

// workspace layout (in _Float16 units)
#define VT_OFF  0
#define VT_SZ   (NB * NH * NW * NCHAN)          // 8388608 halves (16 MiB)
#define CTF_OFF (VT_OFF + VT_SZ)                // ConvTrans frags: 32 pairs * 3t * 32 lanes * 16
#define CTF_SZ  (32 * 3 * 32 * 16)              // 49152
#define ATF_OFF (CTF_OFF + CTF_SZ)              // Attention frags: 8n * 2t * 32 * 16
#define ATF_SZ  (8 * 2 * 32 * 16)               // 8192
#define FEF_OFF (ATF_OFF + ATF_SZ)              // FeaExt frags: 8n * 3t * 32 * 16
#define FEF_SZ  (8 * 3 * 32 * 16)               // 12288
#define WS_TOTAL (FEF_OFF + FEF_SZ)

__device__ __forceinline__ v16h combine16(half8 lo, half8 hi) {
  v16h r;
  #pragma unroll
  for (int i = 0; i < 8; ++i) { r[i] = lo[i]; r[8 + i] = hi[i]; }
  return r;
}

// LDS byte offset of a __shared__ object (addrspace(3) cast -> 32-bit offset)
__device__ __forceinline__ unsigned lds_off(const void* p) {
  return (unsigned)(uintptr_t)(__attribute__((address_space(3))) const void*)p;
}

// ---------------------------------------------------------------------------
// Stage 0: pre-swizzle all WMMA B-fragments into f16 tables in workspace.
// Fragment element i of lane L holds K = t*32 + (L>>4)*16 + i, column L&15.
// ---------------------------------------------------------------------------
__global__ __launch_bounds__(256)
void caps_swizzle(const float* __restrict__ ctW,   // [K=tap*8+d][512]
                  const float* __restrict__ attW,  // [K=d*8+c][64 = n*8+o]
                  const float* __restrict__ feW,   // [K=tap*8+d][64 = n*8+od]
                  _Float16* __restrict__ ws)
{
  const int f = blockIdx.x * 256 + threadIdx.x;
  if (f < CTF_SZ) {
    const int i = f & 15, lane = (f >> 4) & 31, t = (f >> 9) % 3, pair = f / 1536;
    const int c = pair >> 2, j = pair & 3;
    const int K = t * 32 + ((lane >> 4) << 4) + i;
    const int colw = c * 64 + j * 16 + (lane & 15);
    ws[CTF_OFF + f] = (_Float16)((K < 72) ? ctW[(size_t)K * 512 + colw] : 0.f);
  } else if (f < CTF_SZ + ATF_SZ) {
    const int g = f - CTF_SZ;
    const int i = g & 15, lane = (g >> 4) & 31, t = (g >> 9) & 1, n = g >> 10;
    const int K = t * 32 + ((lane >> 4) << 4) + i;
    const int o = lane & 15;
    ws[ATF_OFF + g] = (_Float16)((o < 8) ? attW[(size_t)K * 64 + n * 8 + o] : 0.f);
  } else if (f < CTF_SZ + ATF_SZ + FEF_SZ) {
    const int g = f - (CTF_SZ + ATF_SZ);
    const int i = g & 15, lane = (g >> 4) & 31, t = (g >> 9) % 3, n = g / 1536;
    const int K = t * 32 + ((lane >> 4) << 4) + i;
    const int o = lane & 15;
    ws[FEF_OFF + g] = (_Float16)((K < 72 && o < 8) ? feW[(size_t)K * 64 + n * 8 + o] : 0.f);
  }
}

// ---------------------------------------------------------------------------
// Stage 1: ConvTrans (grouped 3x3, WMMA) -> u (LDS) -> attention (WMMA) ->
//          softmax + weighted sum -> v written as f16 "vt" [B,H,W, n*8+d]
// One workgroup = one (b,h) row of 64 pixels; wave w owns group c = w.
// ---------------------------------------------------------------------------
__global__ __launch_bounds__(256)
void caps_stage1(const float* __restrict__ xin,   // inputs [B,H,W,D,C]
                 const float* __restrict__ ctB,   // [512]
                 const _Float16* __restrict__ ws) // frag tables + vt out
{
  __shared__ __align__(16) _Float16 patch[3 * PC * NCHAN];  // 25344 B
  __shared__ __align__(16) _Float16 u_lds[64 * 8 * 64];     // [px][n][d*8+c] 65536 B
  __shared__ float att_lds[64 * 8 * 8];                     // [px][n][o] 16384 B

  const int bh = blockIdx.x;
  const int b = bh / NH, h = bh % NH;
  const int tid = threadIdx.x;
  const int lane = tid & 31, wave = tid >> 5;
  const int lo16 = lane & 15, hi16 = lane >> 4;
  const _Float16* ctf = ws + CTF_OFF;
  const _Float16* atf = ws + ATF_OFF;
  _Float16* vt = (_Float16*)ws + VT_OFF;

  // ---- stage 3 input rows; per (row,col) load all 64 ch as b128s,
  //      transpose (d,c) -> channel c*8+d in registers, b128 LDS stores
  const size_t inB = (size_t)b * NH * NW * NCHAN;
  for (int i = tid; i < 3 * PC; i += 256) {
    const int col = i % PC, r = i / PC;
    const int hh = h - 1 + r, ww = col - 1;
    _Float16 tmp[64];
    if (hh >= 0 && hh < NH && ww >= 0 && ww < NW) {
      const float* src = &xin[inB + ((size_t)hh * NW + ww) * NCHAN];
      #pragma unroll
      for (int d = 0; d < 8; ++d) {
        f4 v0 = *(const f4*)(src + d * 8);
        f4 v1 = *(const f4*)(src + d * 8 + 4);
        #pragma unroll
        for (int c = 0; c < 4; ++c) tmp[c * 8 + d] = (_Float16)v0[c];
        #pragma unroll
        for (int c = 0; c < 4; ++c) tmp[(c + 4) * 8 + d] = (_Float16)v1[c];
      }
    } else {
      #pragma unroll
      for (int k = 0; k < 64; ++k) tmp[k] = (_Float16)0.f;
    }
    #pragma unroll
    for (int k = 0; k < 8; ++k)
      *(half8*)&patch[i * NCHAN + k * 8] = *(const half8*)&tmp[k * 8];
  }
  __syncthreads();

  // ---- ConvTrans: wave = group c; all 12 B-frags resident, A loaded once/m
  {
    const int c = wave;
    v16h Bf[4][3];
    float bias[4];
    #pragma unroll
    for (int j = 0; j < 4; ++j) {
      #pragma unroll
      for (int t = 0; t < 3; ++t) {
        const _Float16* fb = &ctf[((((c * 4 + j) * 3) + t) * 32 + lane) * 16];
        Bf[j][t] = combine16(*(const half8*)fb, *(const half8*)(fb + 8));
      }
      bias[j] = ctB[c * 64 + j * 16 + lo16];
    }

    #pragma unroll 1
    for (int m = 0; m < 4; ++m) {
      v16h Af[3];
      #pragma unroll
      for (int t = 0; t < 3; ++t) {
        const int pxl  = m * 16 + lo16;
        const int tap0 = t * 4 + hi16 * 2;
        const int tap1 = tap0 + 1;
        half8 a0 = {}, a1 = {};
        if (tap0 < 9) {
          int kr = tap0 / 3, kc = tap0 % 3;
          a0 = *(const half8*)&patch[((kr * PC) + pxl + kc) * NCHAN + c * 8];
        }
        if (tap1 < 9) {
          int kr = tap1 / 3, kc = tap1 % 3;
          a1 = *(const half8*)&patch[((kr * PC) + pxl + kc) * NCHAN + c * 8];
        }
        Af[t] = combine16(a0, a1);
      }
      #pragma unroll
      for (int j = 0; j < 4; ++j) {
        v8f acc = {};
        #pragma unroll
        for (int t = 0; t < 3; ++t)
          acc = __builtin_amdgcn_wmma_f32_16x16x32_f16(false, Af[t], false, Bf[j][t],
                                                       (short)0, acc, false, false);
        // scatter y -> u_lds[px][n][d*8+c]; in-group channel oc = d*8+n
        const int oc = j * 16 + lo16;
        const int d = oc >> 3, n = oc & 7;
        #pragma unroll
        for (int r = 0; r < 8; ++r) {
          const int px = m * 16 + hi16 * 8 + r;
          u_lds[((px * 8) + n) * 64 + d * 8 + c] = (_Float16)(acc[r] + bias[j]);
        }
      }
    }
  }
  __syncthreads();

  // ---- attention logits: att[px, o, n] = sum_{K=(d,c)} u[px,K,n] * A[K,n,o]
  {
    const int n = wave;
    v16h Bfa[2];
    #pragma unroll
    for (int t = 0; t < 2; ++t) {
      const _Float16* fb = &atf[(((n * 2) + t) * 32 + lane) * 16];
      Bfa[t] = combine16(*(const half8*)fb, *(const half8*)(fb + 8));
    }
    #pragma unroll 1
    for (int m = 0; m < 4; ++m) {
      v8f acc = {};
      #pragma unroll
      for (int t = 0; t < 2; ++t) {
        const int pxl = m * 16 + lo16;
        const int kb  = t * 32 + hi16 * 16;
        const _Float16* up = &u_lds[((pxl * 8) + n) * 64 + kb];
        v16h Af = combine16(*(const half8*)up, *(const half8*)(up + 8));
        acc = __builtin_amdgcn_wmma_f32_16x16x32_f16(false, Af, false, Bfa[t],
                                                     (short)0, acc, false, false);
      }
      if (lo16 < 8) {
        #pragma unroll
        for (int r = 0; r < 8; ++r) {
          const int px = m * 16 + hi16 * 8 + r;
          att_lds[(px * 8 + n) * 8 + lo16] = acc[r];
        }
      }
    }
  }
  __syncthreads();

  // ---- softmax over o (8) + v[d] = sum_o u[d,o]*att[o]; store vt f16
  const size_t vtB = ((size_t)b * NH + h) * NW * NCHAN;
  #pragma unroll
  for (int it = 0; it < 2; ++it) {
    const int idx = tid + it * 256;       // (px, n), 512 pairs
    const int px = idx >> 3, n = idx & 7;
    float a[8];
    float mx = -1e30f;
    #pragma unroll
    for (int o = 0; o < 8; ++o) { a[o] = att_lds[(px * 8 + n) * 8 + o]; mx = fmaxf(mx, a[o]); }
    float s = 0.f;
    #pragma unroll
    for (int o = 0; o < 8; ++o) { a[o] = __expf(a[o] - mx); s += a[o]; }
    const float inv = 1.0f / s;
    const _Float16* up = &u_lds[(px * 8 + n) * 64];
    half8 vv;
    #pragma unroll
    for (int d = 0; d < 8; ++d) {
      float acc = 0.f;
      #pragma unroll
      for (int o = 0; o < 8; ++o) acc += (float)up[d * 8 + o] * a[o];
      vv[d] = (_Float16)(acc * inv);
    }
    *(half8*)&vt[vtB + (size_t)px * NCHAN + n * 8] = vv;
  }
}

// ---------------------------------------------------------------------------
// Stage 2: FeaExt (grouped 3x3, WMMA) + bias + ReLU -> CapsAct (1x1, VALU)
// vt staging uses async global->LDS (ASYNCcnt) for the valid interior.
// ---------------------------------------------------------------------------
__global__ __launch_bounds__(256)
void caps_stage2(const _Float16* __restrict__ ws,  // vt + frag tables
                 const float* __restrict__ feB,    // [64]
                 const float* __restrict__ caW,    // [d][64 = n*8+od]
                 const float* __restrict__ caB,    // [64]
                 float* __restrict__ out)          // [B,H,W, d2*8+n] f32
{
  __shared__ __align__(16) _Float16 patch[3 * PC * NCHAN];  // 25344 B
  __shared__ float h1[64 * 64];                             // [px][n*8+od] 16384 B
  __shared__ __align__(16) float o_lds[64 * 64];            // [px][d2*8+n] 16384 B

  const int bh = blockIdx.x;
  const int b = bh / NH, h = bh % NH;
  const int tid = threadIdx.x;
  const int lane = tid & 31, wave = tid >> 5;
  const int lo16 = lane & 15, hi16 = lane >> 4;
  const _Float16* vt  = ws + VT_OFF;
  const _Float16* fef = ws + FEF_OFF;

  // ---- zero the halo / out-of-range portions of the patch
  for (int i = tid; i < 3 * PC * 8; i += 256) {
    const int ch = (i & 7) * 8;
    const int colr = i >> 3;
    const int col = colr % PC, r = colr / PC;
    const int hh = h - 1 + r, ww = col - 1;
    if (!(hh >= 0 && hh < NH && ww >= 0 && ww < NW)) {
      half8 z = {};
      *(half8*)&patch[(r * PC + col) * NCHAN + ch] = z;
    }
  }
  // ---- async copy of the valid interior: LDS[patch] <= vt (b128 per lane)
  for (int i = tid; i < 3 * NW * 8; i += 256) {
    const int ch = (i & 7) * 8;
    const int colr = i >> 3;
    const int ww = colr % NW, r = colr / NW;
    const int hh = h - 1 + r;
    if (hh >= 0 && hh < NH) {
      unsigned ldst = lds_off(&patch[(r * PC + (ww + 1)) * NCHAN + ch]);
      unsigned long long gsrc =
          (unsigned long long)(uintptr_t)&vt[(((size_t)b * NH + hh) * NW + ww) * NCHAN + ch];
      asm volatile("global_load_async_to_lds_b128 %0, %1, off"
                   :: "v"(ldst), "v"(gsrc) : "memory");
    }
  }
  asm volatile("s_wait_asynccnt 0x0" ::: "memory");
  __syncthreads();

  // ---- FeaExt: wave = output type n; B-frags hoisted, A loaded once per m
  {
    const int n = wave;
    v16h Bf[3];
    #pragma unroll
    for (int t = 0; t < 3; ++t) {
      const _Float16* fb = &fef[(((n * 3) + t) * 32 + lane) * 16];
      Bf[t] = combine16(*(const half8*)fb, *(const half8*)(fb + 8));
    }
    const float bias = feB[n * 8 + lo16];

    #pragma unroll 1
    for (int m = 0; m < 4; ++m) {
      v8f acc = {};
      #pragma unroll
      for (int t = 0; t < 3; ++t) {
        const int pxl  = m * 16 + lo16;
        const int tap0 = t * 4 + hi16 * 2, tap1 = tap0 + 1;
        half8 a0 = {}, a1 = {};
        if (tap0 < 9) {
          int kr = tap0 / 3, kc = tap0 % 3;
          a0 = *(const half8*)&patch[((kr * PC) + pxl + kc) * NCHAN + n * 8];
        }
        if (tap1 < 9) {
          int kr = tap1 / 3, kc = tap1 % 3;
          a1 = *(const half8*)&patch[((kr * PC) + pxl + kc) * NCHAN + n * 8];
        }
        v16h Af = combine16(a0, a1);
        acc = __builtin_amdgcn_wmma_f32_16x16x32_f16(false, Af, false, Bf[t],
                                                     (short)0, acc, false, false);
      }
      if (lo16 < 8) {
        #pragma unroll
        for (int r = 0; r < 8; ++r) {
          const int px = m * 16 + hi16 * 8 + r;
          const float hv = acc[r] + bias;
          h1[px * 64 + n * 8 + lo16] = hv > 0.f ? hv : 0.f;
        }
      }
    }
  }
  __syncthreads();

  // ---- CapsAct 1x1 grouped (K=8) in VALU; write to LDS in output layout
  #pragma unroll
  for (int it = 0; it < 2; ++it) {
    const int idx = tid + it * 256;      // (px, n)
    const int px = idx >> 3, n = idx & 7;
    float hv[8];
    #pragma unroll
    for (int d = 0; d < 8; ++d) hv[d] = h1[px * 64 + n * 8 + d];
    #pragma unroll
    for (int od = 0; od < 8; ++od) {
      float s = caB[n * 8 + od];
      #pragma unroll
      for (int d = 0; d < 8; ++d) s += hv[d] * caW[d * 64 + n * 8 + od];
      o_lds[px * 64 + od * 8 + n] = s;
    }
  }
  __syncthreads();

  // ---- coalesced b128 stores of the whole row tile
  const size_t outB = ((size_t)b * NH + h) * NW * NCHAN;
  #pragma unroll
  for (int i = tid; i < 1024; i += 256)
    *(f4*)&out[outB + (size_t)i * 4] = *(const f4*)&o_lds[i * 4];
}

extern "C" void kernel_launch(void* const* d_in, const int* in_sizes, int n_in,
                              void* d_out, int out_size, void* d_ws, size_t ws_size,
                              hipStream_t stream) {
  (void)in_sizes; (void)n_in; (void)out_size; (void)ws_size;
  const float* inputs = (const float*)d_in[0];
  const float* AttW   = (const float*)d_in[1];
  const float* CtW    = (const float*)d_in[2];
  const float* CtB    = (const float*)d_in[3];
  const float* FeW    = (const float*)d_in[4];
  const float* FeB    = (const float*)d_in[5];
  const float* CaW    = (const float*)d_in[6];
  const float* CaB    = (const float*)d_in[7];
  float* out = (float*)d_out;
  _Float16* ws = (_Float16*)d_ws;   // ~16.9 MiB used (vt + swizzled weight frags)

  const int swz_elems = CTF_SZ + ATF_SZ + FEF_SZ;
  caps_swizzle<<<dim3((swz_elems + 255) / 256), dim3(256), 0, stream>>>(CtW, AttW, FeW, ws);
  caps_stage1<<<dim3(NB * NH), dim3(256), 0, stream>>>(inputs, CtB, ws);
  caps_stage2<<<dim3(NB * NH), dim3(256), 0, stream>>>(ws, FeB, CaW, CaB, out);
}
